// EuclCodebook_25159918420254
// MI455X (gfx1250) — matrix operand order repaired and checked by reading, not compiled, not run on hardware
//
#include <hip/hip_runtime.h>
#include <hip/hip_bf16.h>

// ---------------------------------------------------------------------------
// VQ codebook quantization for MI455X (gfx1250, wave32, WMMA).
//   argmin_k ( ||c_k||^2 - 2 * z_row . c_k )   (||z||^2 constant per row)
// GEMM (32768 x 8192 x 256) in bf16 via v_wmma_f32_16x16x32_bf16, f32 accum.
// Double-buffered LDS staging (global prefetch to regs overlaps WMMA block);
// each wave owns 32 rows so one B fragment feeds two WMMAs (1 ds_load_b128
// per WMMA per lane). Loss / z_q / res computed from original fp32 codebook.
// ---------------------------------------------------------------------------

typedef __attribute__((ext_vector_type(16))) __bf16 v16bf;
typedef __attribute__((ext_vector_type(8)))  __bf16 v8bf;
typedef __attribute__((ext_vector_type(4)))  __bf16 v4bf;
typedef __attribute__((ext_vector_type(8)))  float  v8f;

#define EDIM 256          // embedding dim
#define KCODES 8192       // codebook entries
#define NROWS 32768       // B*L tokens
#define BLOCK 128         // 4 waves
#define WAVES 4
#define ROWS_PER_WAVE 32  // two 16-row A tiles per wave
#define ROWS_PER_BLOCK (WAVES * ROWS_PER_WAVE)   // 128
#define NTILE 32          // codes per LDS stage (16 KB)
#define NSTAGES (KCODES / NTILE)                 // 256
#define TILE_U4 (NTILE * EDIM / 8)               // uint4 per tile = 1024

// ---------------------------------------------------------------------------
// Kernel 1: codebook -> bf16 + per-code squared norm (fp32).
// ---------------------------------------------------------------------------
__global__ void vq_prep_codebook(const float* __restrict__ cb,
                                 __bf16* __restrict__ cbb,
                                 float* __restrict__ cnorm) {
    const int row = blockIdx.x;
    const int t   = threadIdx.x;
    const float v = cb[row * EDIM + t];
    cbb[row * EDIM + t] = (__bf16)v;
    float s = v * v;
    #pragma unroll
    for (int m = 16; m >= 1; m >>= 1) s += __shfl_xor(s, m, 32);
    __shared__ float sh[8];
    if ((t & 31) == 0) sh[t >> 5] = s;
    __syncthreads();
    if (t == 0) {
        float tot = 0.f;
        #pragma unroll
        for (int i = 0; i < 8; ++i) tot += sh[i];
        cnorm[row] = tot;
    }
}

// ---------------------------------------------------------------------------
// Kernel 2: z -> bf16 (4 elements / thread, vectorized).
// ---------------------------------------------------------------------------
__global__ void vq_conv_z(const float* __restrict__ z, __bf16* __restrict__ zb) {
    const int i = (blockIdx.x * 256 + threadIdx.x) * 4;
    const float4 v = *(const float4*)(z + i);
    v4bf o;
    o[0] = (__bf16)v.x; o[1] = (__bf16)v.y; o[2] = (__bf16)v.z; o[3] = (__bf16)v.w;
    *(v4bf*)(zb + i) = o;
}

// ---------------------------------------------------------------------------
// Kernel 3: WMMA distance argmin, double-buffered LDS pipeline.
// ---------------------------------------------------------------------------
__launch_bounds__(BLOCK)
__global__ void vq_argmin(const __bf16* __restrict__ zb,
                          const __bf16* __restrict__ cbb,
                          const float*  __restrict__ cnorm,
                          int*          __restrict__ out_idx) {
    __shared__ __bf16 lds_cb[2][NTILE * EDIM];   // 2 x 16 KB
    __shared__ float  lds_cn[2][NTILE];

    const int tid    = threadIdx.x;
    const int wave   = tid >> 5;
    const int lane   = tid & 31;
    const int col    = lane & 15;            // N column / M row selector
    const int half   = lane >> 4;            // 0 or 1
    const int khalfA = half * 8;             // A: K-half offset (ISA layout)
    const int khalfB = half * 8;             // B: K-half offset (ISA layout)

    const int rowBase = blockIdx.x * ROWS_PER_BLOCK + wave * ROWS_PER_WAVE;

    // --- A fragments: 2 row-tiles x 8 K-chunks, resident in VGPRs ---
    v16bf afrag[2][8];
    #pragma unroll
    for (int rt = 0; rt < 2; ++rt) {
        const __bf16* zrow = zb + (size_t)(rowBase + rt * 16 + col) * EDIM;
        #pragma unroll
        for (int kc = 0; kc < 8; ++kc) {
            const __bf16* p = zrow + kc * 32 + khalfA;
            v8bf lo = *(const v8bf*)(p);        // K = base+0..7
            v8bf hi = *(const v8bf*)(p + 16);   // K = base+16..23
            afrag[rt][kc] = __builtin_shufflevector(lo, hi,
                0,1,2,3,4,5,6,7,8,9,10,11,12,13,14,15);
        }
    }

    float best_d[2][8];
    int   best_i[2][8];
    #pragma unroll
    for (int rt = 0; rt < 2; ++rt)
        #pragma unroll
        for (int r = 0; r < 8; ++r) { best_d[rt][r] = 3.4e38f; best_i[rt][r] = 0; }

    const uint4* cbv = (const uint4*)cbb;
    uint4 stage[8];
    float cnreg = 0.f;

    // prologue: tile 0 -> LDS buffer 0
    #pragma unroll
    for (int j = 0; j < 8; ++j) stage[j] = cbv[tid + j * BLOCK];
    if (tid < NTILE) cnreg = cnorm[tid];
    #pragma unroll
    for (int j = 0; j < 8; ++j) ((uint4*)lds_cb[0])[tid + j * BLOCK] = stage[j];
    if (tid < NTILE) lds_cn[0][tid] = cnreg;
    __syncthreads();

    for (int it = 0; it < NSTAGES; ++it) {
        const int cur = it & 1;
        const int nxt = cur ^ 1;
        const int nt  = (it + 1) & (NSTAGES - 1);   // wraps to 0 on last iter

        // prefetch next tile into registers (overlaps the WMMA block)
        {
            const uint4* src = cbv + (size_t)nt * TILE_U4;
            #pragma unroll
            for (int j = 0; j < 8; ++j) stage[j] = src[tid + j * BLOCK];
            if (tid < NTILE) cnreg = cnorm[nt * NTILE + tid];
        }

        // compute on current buffer: 2 code-subtiles x 8 K-chunks x 2 row-tiles
        #pragma unroll
        for (int sub = 0; sub < NTILE / 16; ++sub) {
            const __bf16* brow = &lds_cb[cur][(sub * 16 + col) * EDIM + khalfB];
            v8f acc0 = {};
            v8f acc1 = {};
            #pragma unroll
            for (int kc = 0; kc < 8; ++kc) {
                const __bf16* p = brow + kc * 32;
                v8bf lo = *(const v8bf*)(p);       // K = base+0..7
                v8bf hi = *(const v8bf*)(p + 16);  // K = base+16..23
                v16bf bfrag = __builtin_shufflevector(lo, hi,
                    0,1,2,3,4,5,6,7,8,9,10,11,12,13,14,15);
                acc0 = __builtin_amdgcn_wmma_f32_16x16x32_bf16(
                    false, afrag[0][kc], false, bfrag, (short)0, acc0, false, false);
                acc1 = __builtin_amdgcn_wmma_f32_16x16x32_bf16(
                    false, afrag[1][kc], false, bfrag, (short)0, acc1, false, false);
            }
            const int   code = it * NTILE + sub * 16 + col;
            const float cn   = lds_cn[cur][sub * 16 + col];
            #pragma unroll
            for (int r = 0; r < 8; ++r) {
                const float d0 = __builtin_fmaf(-2.0f, acc0[r], cn);
                if (d0 < best_d[0][r]) { best_d[0][r] = d0; best_i[0][r] = code; }
                const float d1 = __builtin_fmaf(-2.0f, acc1[r], cn);
                if (d1 < best_d[1][r]) { best_d[1][r] = d1; best_i[1][r] = code; }
            }
        }

        // publish prefetched tile into the other buffer
        #pragma unroll
        for (int j = 0; j < 8; ++j) ((uint4*)lds_cb[nxt])[tid + j * BLOCK] = stage[j];
        if (tid < NTILE) lds_cn[nxt][tid] = cnreg;
        __syncthreads();
    }

    // reduce across the 16 column-lanes of each half; tie -> lowest index
    #pragma unroll
    for (int rt = 0; rt < 2; ++rt) {
        #pragma unroll
        for (int r = 0; r < 8; ++r) {
            float d = best_d[rt][r];
            int   i = best_i[rt][r];
            #pragma unroll
            for (int m = 1; m < 16; m <<= 1) {
                float od = __shfl_xor(d, m, 32);
                int   oi = __shfl_xor(i, m, 32);
                if (od < d || (od == d && oi < i)) { d = od; i = oi; }
            }
            if (col == 0) out_idx[rowBase + rt * 16 + r + half * 8] = i;
        }
    }
}

// ---------------------------------------------------------------------------
// Kernel 4: gather z_q (fp32 exact), write z_q_st / res / idx(float),
// per-block partial sums of (z_q - z)^2. One block per token row.
// ---------------------------------------------------------------------------
__global__ void vq_finalize(const float* __restrict__ z,
                            const float* __restrict__ cb,
                            const int*   __restrict__ idx,
                            float* __restrict__ out_zq,
                            float* __restrict__ out_idx_f,
                            float* __restrict__ out_res,
                            float* __restrict__ partials) {
    const int row = blockIdx.x;
    const int e   = threadIdx.x;
    const int i   = row * EDIM + e;
    const int id  = idx[row];
    const float zq = cb[id * EDIM + e];
    const float zv = z[i];
    out_zq[i]  = zq;           // straight-through value == z_q
    out_res[i] = zv - zq;      // z - z_q_st
    if (e == 0) out_idx_f[row] = (float)id;
    const float diff = zq - zv;
    float s = diff * diff;
    #pragma unroll
    for (int m = 16; m >= 1; m >>= 1) s += __shfl_xor(s, m, 32);
    __shared__ float sh[8];
    if ((e & 31) == 0) sh[e >> 5] = s;
    __syncthreads();
    if (e == 0) {
        float tot = 0.f;
        #pragma unroll
        for (int w = 0; w < 8; ++w) tot += sh[w];
        partials[row] = tot;
    }
}

// ---------------------------------------------------------------------------
// Kernel 5: deterministic fixed-order loss reduction (no float atomics).
// loss = mean + COMMIT(=1)*mean = 2*mean(diff^2)
// ---------------------------------------------------------------------------
__global__ void vq_loss_reduce(const float* __restrict__ partials,
                               float* __restrict__ out_loss) {
    const int t = threadIdx.x;
    float s = 0.f;
    for (int k = 0; k < NROWS / 256; ++k) s += partials[t + k * 256];
    #pragma unroll
    for (int m = 16; m >= 1; m >>= 1) s += __shfl_xor(s, m, 32);
    __shared__ float sh[8];
    if ((t & 31) == 0) sh[t >> 5] = s;
    __syncthreads();
    if (t == 0) {
        float tot = 0.f;
        #pragma unroll
        for (int w = 0; w < 8; ++w) tot += sh[w];
        out_loss[0] = 2.0f * tot / (float)(NROWS * EDIM);
    }
}

// ---------------------------------------------------------------------------
// Launch. d_out layout (fp32): [z_q_st 8388608][idx 32768][loss 1][res 8388608]
// Workspace (~21.3 MB): z_bf16 | cb_bf16 | cnorm | idx_i32 | partials
// ---------------------------------------------------------------------------
extern "C" void kernel_launch(void* const* d_in, const int* in_sizes, int n_in,
                              void* d_out, int out_size, void* d_ws, size_t ws_size,
                              hipStream_t stream) {
    const float* z  = (const float*)d_in[0];   // 32*1024*256
    const float* cb = (const float*)d_in[1];   // 8192*256

    float* out       = (float*)d_out;
    float* out_zq    = out;
    float* out_idx_f = out + (size_t)NROWS * EDIM;
    float* out_loss  = out_idx_f + NROWS;
    float* out_res   = out_loss + 1;

    char* ws = (char*)d_ws;
    __bf16* zb       = (__bf16*)(ws);                        // 16,777,216 B
    __bf16* cbb      = (__bf16*)(ws + 16777216);             //  4,194,304 B
    float*  cnorm    = (float*) (ws + 20971520);             //     32,768 B
    int*    idx_i    = (int*)   (ws + 21004288);             //    131,072 B
    float*  partials = (float*) (ws + 21135360);             //    131,072 B

    vq_prep_codebook<<<KCODES, 256, 0, stream>>>(cb, cbb, cnorm);
    vq_conv_z<<<(NROWS * EDIM) / (256 * 4), 256, 0, stream>>>(z, zb);
    vq_argmin<<<NROWS / ROWS_PER_BLOCK, BLOCK, 0, stream>>>(zb, cbb, cnorm, idx_i);
    vq_finalize<<<NROWS, 256, 0, stream>>>(z, cb, idx_i, out_zq, out_idx_f,
                                           out_res, partials);
    vq_loss_reduce<<<1, 256, 0, stream>>>(partials, out_loss);
}
